// DeployLinear_88940182766343
// MI455X (gfx1250) — 8-bit matrix operand order repaired and checked by reading, compiled, not run on hardware
//
#include <hip/hip_runtime.h>
#include <hip/hip_bf16.h>
#include <math.h>
#include <stdint.h>

typedef int v8i __attribute__((ext_vector_type(8)));

#define QMAXF 7.0f
#define QMINF -8.0f
#define EPSF  1e-8f

template<int K>
__global__ __launch_bounds__(256) void fwht_quant_kernel(
    const float* __restrict__ x, signed char* __restrict__ qx,
    float* __restrict__ sx)
{
  __shared__ float buf[K];
  __shared__ float red[256];
  const int row = blockIdx.x;
  const int t   = threadIdx.x;
  const float* xr = x + (size_t)row * K;

  for (int i = t; i < K; i += 256) buf[i] = xr[i];
  __syncthreads();

  for (int h = 1; h < K; h <<= 1) {
    for (int i = t; i < K / 2; i += 256) {
      int lo  = i & (h - 1);
      int pos = ((i - lo) << 1) + lo;
      float a = buf[pos];
      float c = buf[pos + h];
      buf[pos]     = a + c;
      buf[pos + h] = a - c;
    }
    __syncthreads();
  }

  float mx = 0.0f;
  for (int i = t; i < K; i += 256) mx = fmaxf(mx, fabsf(buf[i]));
  red[t] = mx;
  __syncthreads();
  for (int sdim = 128; sdim > 0; sdim >>= 1) {
    if (t < sdim) red[t] = fmaxf(red[t], red[t + sdim]);
    __syncthreads();
  }

  const float sqrtK = sqrtf((float)K);
  float denom = fmaxf(red[0] * (1.0f / QMAXF), EPSF * sqrtK);
  if (t == 0) sx[row] = denom / sqrtK;
  float inv = 1.0f / denom;

  int* qout = (int*)(qx + (size_t)row * K);
  for (int i = t; i < K / 4; i += 256) {
    int pk = 0;
#pragma unroll
    for (int j = 0; j < 4; j++) {
      float q = rintf(buf[i * 4 + j] * inv);
      q = fminf(fmaxf(q, QMINF), QMAXF);
      pk |= (((int)q) & 0xFF) << (8 * j);
    }
    qout[i] = pk;
  }
}

template<int K>
__global__ __launch_bounds__(256) void quant_w_kernel(
    const float* __restrict__ W, signed char* __restrict__ qw,
    float* __restrict__ sw)
{
  __shared__ float red[256];
  const int row = blockIdx.x;
  const int t   = threadIdx.x;
  const float* wr = W + (size_t)row * K;

  float mx = 0.0f;
  for (int i = t; i < K; i += 256) mx = fmaxf(mx, fabsf(wr[i]));
  red[t] = mx;
  __syncthreads();
  for (int sdim = 128; sdim > 0; sdim >>= 1) {
    if (t < sdim) red[t] = fmaxf(red[t], red[t + sdim]);
    __syncthreads();
  }

  float s = fmaxf(red[0] * (1.0f / QMAXF), EPSF);
  if (t == 0) sw[row] = s;
  float inv = 1.0f / s;

  int* qout = (int*)(qw + (size_t)row * K);
  for (int i = t; i < K / 4; i += 256) {
    int pk = 0;
#pragma unroll
    for (int j = 0; j < 4; j++) {
      float q = rintf(wr[i * 4 + j] * inv);
      q = fminf(fmaxf(q, QMINF), QMAXF);
      pk |= (((int)q) & 0xFF) << (8 * j);
    }
    qout[i] = pk;
  }
}

__device__ __forceinline__ void async_copy_b128(void* lds_dst, const void* gsrc) {
  uint32_t lds = (uint32_t)(uintptr_t)lds_dst;
  asm volatile("global_load_async_to_lds_b128 %0, %1, off"
               :: "v"(lds), "v"(gsrc) : "memory");
}

__device__ __forceinline__ void wait_async0() {
#if __has_builtin(__builtin_amdgcn_s_wait_asynccnt)
  __builtin_amdgcn_s_wait_asynccnt(0);
#else
  asm volatile("s_wait_asynccnt 0" ::: "memory");
#endif
}

__global__ __launch_bounds__(256) void gemm_iu8_lds_kernel(
    const signed char* __restrict__ qx, const signed char* __restrict__ qw,
    const float* __restrict__ sx, const float* __restrict__ sw,
    const float* __restrict__ bias, float* __restrict__ out,
    int M, int N, int K)
{
  constexpr int KSTEP = 128;
  constexpr int LROW  = 144;
  __shared__ __align__(16) signed char As[2][128 * LROW];
  __shared__ __align__(16) signed char Bs[2][128 * LROW];

  const int tid  = threadIdx.x;
  const int wave = tid >> 5;
  const int lane = tid & 31;
  const int l16  = lane & 15;
  const int hi   = lane >> 4;
  const int wm   = wave & 3;
  const int wn   = wave >> 2;

  const int blockM = blockIdx.y * 128;
  const int blockN = blockIdx.x * 128;

  const int sr    = tid >> 1;
  const int shalf = (tid & 1) * 64;
  const signed char* gA = qx + (size_t)(blockM + sr) * K + shalf;
  const signed char* gB = qw + (size_t)(blockN + sr) * K + shalf;
  const int sOff = sr * LROW + shalf;

  const int aRow0 = (wm * 32 + l16) * LROW + hi * 16;
  const int bRow0 = (wn * 64 + l16) * LROW + hi * 16;

  v8i acc[2][4] = {};
  const int nstages = K / KSTEP;

#pragma unroll
  for (int q = 0; q < 4; q++) {
    async_copy_b128(&As[0][sOff + q * 16], gA + q * 16);
    async_copy_b128(&Bs[0][sOff + q * 16], gB + q * 16);
  }
  wait_async0();
  __syncthreads();

  for (int stage = 0; stage < nstages; ++stage) {
    const int cur  = stage & 1;
    const bool more = (stage + 1 < nstages);

    if (more) {
      const int kk  = (stage + 1) * KSTEP;
      const int nxt = cur ^ 1;
#pragma unroll
      for (int q = 0; q < 4; q++) {
        async_copy_b128(&As[nxt][sOff + q * 16], gA + kk + q * 16);
        async_copy_b128(&Bs[nxt][sOff + q * 16], gB + kk + q * 16);
      }
    }

#pragma unroll
    for (int s = 0; s < 2; ++s) {
      const int ks = s * 64;
      v8i a[2], bf[4];
#pragma unroll
      for (int i = 0; i < 2; i++) {
        const signed char* p = &As[cur][aRow0 + i * 16 * LROW + ks];
        int4 p03 = *(const int4*)(p);
        int4 p47 = *(const int4*)(p + 32);
        a[i][0] = p03.x; a[i][1] = p03.y; a[i][2] = p03.z; a[i][3] = p03.w;
        a[i][4] = p47.x; a[i][5] = p47.y; a[i][6] = p47.z; a[i][7] = p47.w;
      }
#pragma unroll
      for (int j = 0; j < 4; j++) {
        const signed char* p = &Bs[cur][bRow0 + j * 16 * LROW + ks];
        int4 q03 = *(const int4*)(p);
        int4 q47 = *(const int4*)(p + 32);
        bf[j][0] = q03.x; bf[j][1] = q03.y; bf[j][2] = q03.z; bf[j][3] = q03.w;
        bf[j][4] = q47.x; bf[j][5] = q47.y; bf[j][6] = q47.z; bf[j][7] = q47.w;
      }
#pragma unroll
      for (int i = 0; i < 2; i++)
#pragma unroll
        for (int j = 0; j < 4; j++)
          acc[i][j] = __builtin_amdgcn_wmma_i32_16x16x64_iu8(
               true, a[i],  true, bf[j], acc[i][j],
               false,  false);
    }

    if (more) {
      wait_async0();
      __syncthreads();
    }
  }

  const int m0 = blockM + wm * 32;
  const int n0 = blockN + wn * 64;
#pragma unroll
  for (int i = 0; i < 2; i++) {
    float sxv[8];
#pragma unroll
    for (int g = 0; g < 8; g++) sxv[g] = sx[m0 + i * 16 + hi * 8 + g];
#pragma unroll
    for (int j = 0; j < 4; j++) {
      int n     = n0 + j * 16 + l16;
      float swn = sw[n];
      float bn  = bias[n];
#pragma unroll
      for (int g = 0; g < 8; g++) {
        int m = m0 + i * 16 + hi * 8 + g;
        out[(size_t)m * N + n] = (float)acc[i][j][g] * sxv[g] * swn + bn;
      }
    }
  }
}

extern "C" void kernel_launch(void* const* d_in, const int* in_sizes, int n_in,
                              void* d_out, int out_size, void* d_ws, size_t ws_size,
                              hipStream_t stream)
{
  const float* x = (const float*)d_in[0];
  const float* W = (const float*)d_in[1];
  const float* b = (const float*)d_in[2];
  float* out = (float*)d_out;

  const int N = in_sizes[2];
  const int K = in_sizes[1] / N;
  const int M = in_sizes[0] / K;

  signed char* qx = (signed char*)d_ws;
  signed char* qw = qx + (size_t)M * K;
  float* sx = (float*)(qw + (size_t)N * K);
  float* sw = sx + M;

  fwht_quant_kernel<2048><<<M, 256, 0, stream>>>(x, qx, sx);
  quant_w_kernel<2048><<<N, 256, 0, stream>>>(W, qw, sw);

  dim3 grid(N / 128, M / 128);
  gemm_iu8_lds_kernel<<<grid, 256, 0, stream>>>(qx, qw, sx, sw, b, out, M, N, K);
}
